// ContrastiveLearningLoss_80255758893848
// MI455X (gfx1250) — matrix operand order repaired and checked by reading, not compiled, not run on hardware
//
#include <hip/hip_runtime.h>
#include <hip/hip_bf16.h>
#include <math.h>

typedef __attribute__((ext_vector_type(16))) _Float16 v16h;
typedef __attribute__((ext_vector_type(8)))  _Float16 v8h;
typedef __attribute__((ext_vector_type(8)))  float    v8f;

#define DIMS    256
#define NHALF   4096
#define NROWS   8192   // 2*NHALF
#define KCH     8      // 256 / 32
#define CTILES  512    // 8192 / 16
#define NSLABS  8      // column slabs (parallel partial row-sums)
#define TPS     (CTILES / NSLABS)   // 64 column tiles per slab
#define LOG2E2  2.885390081777926814f  // 2 * log2(e): exp(2x) = exp2(x*LOG2E2)

// ---------------------------------------------------------------------------
// Kernel 1: L2-normalize rows of concat(z_i, z_j); emit fp32 + f16 copies.
// One wave per row (8 floats per lane), 8 rows per 256-thread block.
// ---------------------------------------------------------------------------
__global__ __launch_bounds__(256) void nrm_kernel(const float* __restrict__ zi,
                                                  const float* __restrict__ zj,
                                                  float* __restrict__ repsF,
                                                  _Float16* __restrict__ repsH) {
  const int row  = blockIdx.x * 8 + (threadIdx.x >> 5);
  const int lane = threadIdx.x & 31;
  const float* src = (row < NHALF) ? (zi + (size_t)row * DIMS)
                                   : (zj + (size_t)(row - NHALF) * DIMS);
  const float4* s4 = (const float4*)src;
  float4 a = s4[lane * 2 + 0];
  float4 b = s4[lane * 2 + 1];
  float ss = a.x*a.x + a.y*a.y + a.z*a.z + a.w*a.w
           + b.x*b.x + b.y*b.y + b.z*b.z + b.w*b.w;
  ss += __shfl_xor(ss, 1,  32);
  ss += __shfl_xor(ss, 2,  32);
  ss += __shfl_xor(ss, 4,  32);
  ss += __shfl_xor(ss, 8,  32);
  ss += __shfl_xor(ss, 16, 32);
  const float inv = 1.0f / fmaxf(sqrtf(ss), 1e-12f);
  a.x *= inv; a.y *= inv; a.z *= inv; a.w *= inv;
  b.x *= inv; b.y *= inv; b.z *= inv; b.w *= inv;
  float4* dF = (float4*)(repsF + (size_t)row * DIMS);
  dF[lane * 2 + 0] = a;
  dF[lane * 2 + 1] = b;
  v8h h;
  h[0]=(_Float16)a.x; h[1]=(_Float16)a.y; h[2]=(_Float16)a.z; h[3]=(_Float16)a.w;
  h[4]=(_Float16)b.x; h[5]=(_Float16)b.y; h[6]=(_Float16)b.z; h[7]=(_Float16)b.w;
  *(v8h*)(repsH + (size_t)row * DIMS + lane * 8) = h;
}

// ---------------------------------------------------------------------------
// Kernel 2: fused sim-GEMM + per-slab partial row sums of exp(2*dot).
// Grid: (64 row blocks) x (8 column slabs). Wave = 16 rows, A register-
// resident; all 8 B chunks batched before the WMMA burst so the backend can
// clause the loads and overlap with compute.
// ---------------------------------------------------------------------------
__global__ __launch_bounds__(256) void simrow_kernel(const _Float16* __restrict__ repsH,
                                                     float* __restrict__ rowsumP) {
  const int lane    = threadIdx.x & 31;
  const int wave    = threadIdx.x >> 5;
  const int rowbase = blockIdx.x * 128 + wave * 16;
  const int slab    = blockIdx.y;
  const int m = lane & 15;   // row (A) / column (B,C) within tile
  const int h = lane >> 4;   // half-wave selector

  // --- A fragments: 8 K-chunks of the wave's 16x256 row block (contiguous K order, k = 16h + s, same as B) ---
  union { v16h v; v8h p[2]; } a[KCH];
  const _Float16* arow = repsH + (size_t)(rowbase + m) * DIMS;
  #pragma unroll
  for (int kc = 0; kc < KCH; ++kc) {
    const _Float16* p = arow + kc * 32 + h * 16;
    a[kc].p[0] = *(const v8h*)(p);        // K = kc*32 + h*16 .. +7
    a[kc].p[1] = *(const v8h*)(p + 8);    // K = kc*32 + h*16 + 8 .. +15
  }

  float part[8];
  #pragma unroll
  for (int v = 0; v < 8; ++v) part[v] = 0.0f;

  for (int t = 0; t < TPS; ++t) {
    const int ct = slab * TPS + t;
    const _Float16* bcol = repsH + (size_t)(ct * 16 + m) * DIMS + h * 16;
    if (ct + 1 < CTILES)   // hint next tile into cache (global_prefetch_b8)
      __builtin_prefetch(bcol + 16 * DIMS, 0, 1);

    // Batch all B chunk loads -> one clause, partial waits against WMMAs.
    v16h b[KCH];
    #pragma unroll
    for (int kc = 0; kc < KCH; ++kc)
      b[kc] = *(const v16h*)(bcol + kc * 32);   // 16 contiguous K for col m

    v8f c = {};
    #pragma unroll
    for (int kc = 0; kc < KCH; ++kc)
      c = __builtin_amdgcn_wmma_f32_16x16x32_f16(
              /*neg_a=*/false, a[kc].v, /*neg_b=*/false, b[kc],
              /*c_mod=*/(short)0, c, /*reuse_a=*/false, /*reuse_b=*/false);

    #pragma unroll
    for (int v = 0; v < 8; ++v)
      part[v] += __builtin_amdgcn_exp2f(c[v] * LOG2E2);  // exp(sim/T), T=0.5
  }

  // Reduce each row's partials across its 16-lane half; store slab partial.
  #pragma unroll
  for (int v = 0; v < 8; ++v) {
    float s = part[v];
    s += __shfl_xor(s, 1, 32);
    s += __shfl_xor(s, 2, 32);
    s += __shfl_xor(s, 4, 32);
    s += __shfl_xor(s, 8, 32);
    if (m == 0)
      rowsumP[(size_t)slab * NROWS + rowbase + h * 8 + v] = s;
  }
}

// ---------------------------------------------------------------------------
// Kernel 3: per-row contribution  log(S_r) - 2*dot(reps[r], reps[label_r]).
// One wave per row on fp32 normalized reps; folds the 8 slab partials.
// ---------------------------------------------------------------------------
__global__ __launch_bounds__(256) void tgt_kernel(const float* __restrict__ repsF,
                                                  const float* __restrict__ rowsumP,
                                                  float* __restrict__ contrib) {
  const int row  = blockIdx.x * 8 + (threadIdx.x >> 5);
  const int lane = threadIdx.x & 31;
  const int lbl  = (row < NHALF) ? (2 * row) : (2 * (row - NHALF) + 1);
  const float4* pr = (const float4*)(repsF + (size_t)row * DIMS);
  const float4* pl = (const float4*)(repsF + (size_t)lbl * DIMS);
  float d = 0.0f;
  #pragma unroll
  for (int i = 0; i < 2; ++i) {
    float4 x = pr[lane * 2 + i];
    float4 y = pl[lane * 2 + i];
    d += x.x*y.x + x.y*y.y + x.z*y.z + x.w*y.w;
  }
  d += __shfl_xor(d, 1,  32);
  d += __shfl_xor(d, 2,  32);
  d += __shfl_xor(d, 4,  32);
  d += __shfl_xor(d, 8,  32);
  d += __shfl_xor(d, 16, 32);
  if (lane == 0) {
    float S = 0.0f;
    #pragma unroll
    for (int s = 0; s < NSLABS; ++s)
      S += rowsumP[(size_t)s * NROWS + row];
    contrib[row] = logf(S) - 2.0f * d;
  }
}

// ---------------------------------------------------------------------------
// Kernel 4: mean over 8192 contributions -> scalar loss.
// ---------------------------------------------------------------------------
__global__ __launch_bounds__(256) void red_kernel(const float* __restrict__ contrib,
                                                  float* __restrict__ out) {
  __shared__ float sm[256];
  float s = 0.0f;
  for (int i = threadIdx.x; i < NROWS; i += 256) s += contrib[i];
  sm[threadIdx.x] = s;
  __syncthreads();
  for (int off = 128; off > 0; off >>= 1) {
    if ((int)threadIdx.x < off) sm[threadIdx.x] += sm[threadIdx.x + off];
    __syncthreads();
  }
  if (threadIdx.x == 0) out[0] = sm[0] / (float)NROWS;
}

// ---------------------------------------------------------------------------
extern "C" void kernel_launch(void* const* d_in, const int* in_sizes, int n_in,
                              void* d_out, int out_size, void* d_ws, size_t ws_size,
                              hipStream_t stream) {
  const float* zi = (const float*)d_in[0];
  const float* zj = (const float*)d_in[1];
  float* out = (float*)d_out;

  // Workspace layout
  char* ws = (char*)d_ws;
  float*    repsF   = (float*)ws;                                   // 8 MB
  _Float16* repsH   = (_Float16*)(ws + (size_t)NROWS * DIMS * 4);   // 4 MB
  float*    rowsumP = (float*)(ws + (size_t)NROWS * DIMS * 6);      // 256 KB
  float*    contrib = rowsumP + (size_t)NSLABS * NROWS;             // 32 KB

  nrm_kernel<<<NROWS / 8, 256, 0, stream>>>(zi, zj, repsF, repsH);
  dim3 g2(NROWS / 128, NSLABS);
  simrow_kernel<<<g2, 256, 0, stream>>>(repsH, rowsumP);
  tgt_kernel<<<NROWS / 8, 256, 0, stream>>>(repsF, rowsumP, contrib);
  red_kernel<<<1, 256, 0, stream>>>(contrib, out);
}
